// NeighborEmbedding_50577534877741
// MI455X (gfx1250) — matrix operand order repaired and checked by reading, not compiled, not run on hardware
//
#include <hip/hip_runtime.h>
#include <hip/hip_bf16.h>

typedef __attribute__((ext_vector_type(16))) __bf16   bf16x16;
typedef __attribute__((ext_vector_type(8)))  __bf16   bf16x8;
typedef __attribute__((ext_vector_type(2)))  __bf16   bf16x2;
typedef __attribute__((ext_vector_type(8)))  float    f32x8;

#define LAMDA 0.8f

// Hardware bf16 convert (gfx1250 native) + pair pack
__device__ __forceinline__ unsigned packbf2(float a, float b) {
    bf16x2 v;
    v[0] = (__bf16)a;
    v[1] = (__bf16)b;
    return __builtin_bit_cast(unsigned, v);
}

// ---------------- GEMM: H[V,D] = bf16(embedding[V,K]) @ bf16(W[K,D]) + b ----
// Block: 256 threads (8 wave32), tile 64(M) x 128(N), BK=32.
// A tiles: async global->LDS DMA (f32, double buffered), convert on fragment build.
// B tiles: VGPR-staged, packed bf16 stored transposed [col][k] in LDS.
#define BM 64
#define BN 128
#define BK 32
#define AFSTR 36   // f32 per A row in LDS (128B + 16B pad -> conflict-free b128)
#define BSU   20   // u32 per B column in LDS (16 data + 4 pad = 80B, 16B aligned)

__global__ __launch_bounds__(256)
void gemm_bf16_wmma(const float* __restrict__ A, const float* __restrict__ Wm,
                    const float* __restrict__ bias, float* __restrict__ H,
                    int K, int D) {
    __shared__ float    Af[2][BM * AFSTR];   // f32 A tiles, async-filled
    __shared__ unsigned Bs[2][BN * BSU];     // bf16-pair B tiles, [col][k]

    const int t    = threadIdx.x;
    const int wave = t >> 5;
    const int lane = t & 31;
    const int m0   = blockIdx.x * BM;

    const int rowblk  = wave & 3;   // which 16-row block of the 64
    const int colhalf = wave >> 2;  // which 64-col half of the 128

    // Per-thread staging geometry
    const int arow  = t >> 2;              // A: 64 rows, 4 threads/row
    const int acol  = (t & 3) * 8;         // 8 consecutive f32 (2 x b128)
    const int brow2 = t >> 5;              // B it=0: k-pair 0..7
    const int bcol  = (t & 31) * 4;        // 4 consecutive columns

    f32x8 acc[4] = {};
    float4 breg[4];

    const int nsteps = K / BK;

    // ---- prologue: stage tile 0 into buffer 0 ----
    {
        const float* g = A + (size_t)(m0 + arow) * K + acol;
        unsigned l = (unsigned)(uintptr_t)&Af[0][arow * AFSTR + acol];
        unsigned long long g0 = (unsigned long long)(uintptr_t)g;
        unsigned long long g1 = g0 + 16ull;
        asm volatile("global_load_async_to_lds_b128 %0, %1, off"
                     :: "v"(l), "v"(g0) : "memory");
        asm volatile("global_load_async_to_lds_b128 %0, %1, off"
                     :: "v"(l + 16u), "v"(g1) : "memory");

        #pragma unroll
        for (int it = 0; it < 2; ++it) {
            int r2 = brow2 + it * 8;
            breg[it * 2 + 0] = *(const float4*)(Wm + (size_t)(2 * r2 + 0) * D + bcol);
            breg[it * 2 + 1] = *(const float4*)(Wm + (size_t)(2 * r2 + 1) * D + bcol);
        }
        #pragma unroll
        for (int it = 0; it < 2; ++it) {
            int r2 = brow2 + it * 8;
            float4 lo = breg[it * 2 + 0], hi = breg[it * 2 + 1];
            Bs[0][(bcol + 0) * BSU + r2] = packbf2(lo.x, hi.x);
            Bs[0][(bcol + 1) * BSU + r2] = packbf2(lo.y, hi.y);
            Bs[0][(bcol + 2) * BSU + r2] = packbf2(lo.z, hi.z);
            Bs[0][(bcol + 3) * BSU + r2] = packbf2(lo.w, hi.w);
        }
        asm volatile("s_wait_asynccnt 0x0" ::: "memory");
    }
    __syncthreads();

    for (int s = 0; s < nsteps; ++s) {
        const int cur = s & 1;
        const int nxt = cur ^ 1;
        const bool hasNext = (s + 1) < nsteps;
        const int k0n = (s + 1) * BK;

        // ---- issue next-tile staging (A async DMA, B to VGPRs) ----
        if (hasNext) {
            const float* g = A + (size_t)(m0 + arow) * K + k0n + acol;
            unsigned l = (unsigned)(uintptr_t)&Af[nxt][arow * AFSTR + acol];
            unsigned long long g0 = (unsigned long long)(uintptr_t)g;
            unsigned long long g1 = g0 + 16ull;
            asm volatile("global_load_async_to_lds_b128 %0, %1, off"
                         :: "v"(l), "v"(g0) : "memory");
            asm volatile("global_load_async_to_lds_b128 %0, %1, off"
                         :: "v"(l + 16u), "v"(g1) : "memory");
            #pragma unroll
            for (int it = 0; it < 2; ++it) {
                int r2 = brow2 + it * 8;
                breg[it * 2 + 0] = *(const float4*)(Wm + (size_t)(k0n + 2 * r2 + 0) * D + bcol);
                breg[it * 2 + 1] = *(const float4*)(Wm + (size_t)(k0n + 2 * r2 + 1) * D + bcol);
            }
        }

        // ---- compute from current buffers ----
        // A fragment (ISA 16-bit A 16x32 layout):
        // lanes 0-15: M=lane,    K = {0..7, 16..23}
        // lanes16-31: M=lane-16, K = {8..15, 24..31}
        const int mrow = rowblk * 16 + (lane & 15);
        const int kb   = (lane >> 4) * 8;
        const float4* ap0 = (const float4*)&Af[cur][mrow * AFSTR + kb];
        const float4* ap1 = (const float4*)&Af[cur][mrow * AFSTR + kb + 16];
        float4 a0 = ap0[0], a1 = ap0[1];
        float4 a2 = ap1[0], a3 = ap1[1];
        bf16x16 afrag;
        afrag[0]  = (__bf16)a0.x; afrag[1]  = (__bf16)a0.y;
        afrag[2]  = (__bf16)a0.z; afrag[3]  = (__bf16)a0.w;
        afrag[4]  = (__bf16)a1.x; afrag[5]  = (__bf16)a1.y;
        afrag[6]  = (__bf16)a1.z; afrag[7]  = (__bf16)a1.w;
        afrag[8]  = (__bf16)a2.x; afrag[9]  = (__bf16)a2.y;
        afrag[10] = (__bf16)a2.z; afrag[11] = (__bf16)a2.w;
        afrag[12] = (__bf16)a3.x; afrag[13] = (__bf16)a3.y;
        afrag[14] = (__bf16)a3.z; afrag[15] = (__bf16)a3.w;

        // B fragments: N = lane%16; lanes 0-15 hold K={0..7,16..23}, lanes16-31 K={8..15,24..31}
        const int ku = (lane >> 4) * 4;   // u32 offset into column: 0 or 4
        bf16x16 bfrag[4];
        #pragma unroll
        for (int n = 0; n < 4; ++n) {
            const int col = (colhalf * 4 + n) * 16 + (lane & 15);
            const unsigned* colp = &Bs[cur][col * BSU + ku];
            bf16x8 b0 = *reinterpret_cast<const bf16x8*>(colp);
            bf16x8 b1 = *reinterpret_cast<const bf16x8*>(colp + 8);
            bfrag[n] = __builtin_shufflevector(b0, b1,
                0,1,2,3,4,5,6,7,8,9,10,11,12,13,14,15);
        }
        #pragma unroll
        for (int n = 0; n < 4; ++n)
            acc[n] = __builtin_amdgcn_wmma_f32_16x16x32_bf16(
                false, afrag, false, bfrag[n], (short)0, acc[n], false, false);

        // ---- convert + store next B tile (lands behind the WMMAs) ----
        if (hasNext) {
            #pragma unroll
            for (int it = 0; it < 2; ++it) {
                int r2 = brow2 + it * 8;
                float4 lo = breg[it * 2 + 0], hi = breg[it * 2 + 1];
                Bs[nxt][(bcol + 0) * BSU + r2] = packbf2(lo.x, hi.x);
                Bs[nxt][(bcol + 1) * BSU + r2] = packbf2(lo.y, hi.y);
                Bs[nxt][(bcol + 2) * BSU + r2] = packbf2(lo.z, hi.z);
                Bs[nxt][(bcol + 3) * BSU + r2] = packbf2(lo.w, hi.w);
            }
        }
        asm volatile("s_wait_asynccnt 0x0" ::: "memory");
        __syncthreads();
    }

    // C/D layout: VGPR v, lanes 0-15 -> M=v, lanes 16-31 -> M=v+8; N=lane%16
    const int mbase = m0 + rowblk * 16 + ((lane >> 4) << 3);
    #pragma unroll
    for (int n = 0; n < 4; ++n) {
        const int col = (colhalf * 4 + n) * 16 + (lane & 15);
        const float bv = bias[col];
        #pragma unroll
        for (int v = 0; v < 8; ++v)
            H[(size_t)(mbase + v) * D + col] = acc[n][v] + bv;
    }
}

// ---------------- Edge scatter: P[dst] += w * H[src] ------------------------
__global__ __launch_bounds__(256)
void scatter_edges(const int* __restrict__ src, const int* __restrict__ dst,
                   const float* __restrict__ w, const float* __restrict__ H,
                   float* __restrict__ P, int E, int D) {
    int gid = blockIdx.x * blockDim.x + threadIdx.x;
    int e   = gid >> 5;                 // one wave32 per edge
    if (e >= E) return;
    int lane = threadIdx.x & 31;
    int s = src[e], d = dst[e];
    float ww = w[e];
    float4 m = ((const float4*)(H + (size_t)s * D))[lane];
    float* p = P + (size_t)d * D + lane * 4;
    unsafeAtomicAdd(p + 0, m.x * ww);
    unsafeAtomicAdd(p + 1, m.y * ww);
    unsafeAtomicAdd(p + 2, m.z * ww);
    unsafeAtomicAdd(p + 3, m.w * ww);
}

// ---------------- Gather + blend + L2-normalize -----------------------------
__global__ __launch_bounds__(256)
void gather_normalize(const int* __restrict__ x, const float* __restrict__ H,
                      const float* __restrict__ P, float* __restrict__ out,
                      int Nrows, int D) {
    int gid = blockIdx.x * blockDim.x + threadIdx.x;
    int r   = gid >> 5;                 // one wave32 per output row
    if (r >= Nrows) return;
    int lane = threadIdx.x & 31;
    int xi = x[r];
    float4 p = ((const float4*)(P + (size_t)xi * D))[lane];
    float4 h = ((const float4*)(H + (size_t)xi * D))[lane];
    float4 e;
    e.x = LAMDA * p.x + (1.0f - LAMDA) * h.x;
    e.y = LAMDA * p.y + (1.0f - LAMDA) * h.y;
    e.z = LAMDA * p.z + (1.0f - LAMDA) * h.z;
    e.w = LAMDA * p.w + (1.0f - LAMDA) * h.w;
    float ss = e.x * e.x + e.y * e.y + e.z * e.z + e.w * e.w;
    #pragma unroll
    for (int off = 16; off > 0; off >>= 1)
        ss += __shfl_xor(ss, off, 32);
    float sc = 1.0f / fmaxf(sqrtf(ss), 1e-12f);
    float4 o = make_float4(e.x * sc, e.y * sc, e.z * sc, e.w * sc);
    ((float4*)(out + (size_t)r * D))[lane] = o;
}

// ---------------- Launch ----------------------------------------------------
extern "C" void kernel_launch(void* const* d_in, const int* in_sizes, int n_in,
                              void* d_out, int out_size, void* d_ws, size_t ws_size,
                              hipStream_t stream) {
    const int*   x    = (const int*)d_in[0];
    const int*   ei   = (const int*)d_in[1];   // [2,E] flat: src then dst
    const float* ew   = (const float*)d_in[2];
    const float* emb  = (const float*)d_in[3]; // [V,V]
    const float* Wm   = (const float*)d_in[4]; // [V,D]
    const float* bias = (const float*)d_in[5]; // [D]

    const int Nrows = in_sizes[0];
    const int E     = in_sizes[2];
    const int D     = in_sizes[5];             // 128
    const int V     = in_sizes[4] / D;         // 8192

    float* H = (float*)d_ws;                   // [V,D]
    float* P = H + (size_t)V * D;              // [V,D]

    hipMemsetAsync(P, 0, (size_t)V * D * sizeof(float), stream);

    gemm_bf16_wmma<<<V / BM, 256, 0, stream>>>(emb, Wm, bias, H, V, D);

    int blk = 256;
    int edgeThreads = E * 32;
    scatter_edges<<<(edgeThreads + blk - 1) / blk, blk, 0, stream>>>(
        ei, ei + E, ew, H, P, E, D);

    int rowThreads = Nrows * 32;
    gather_normalize<<<(rowThreads + blk - 1) / blk, blk, 0, stream>>>(
        x, H, P, (float*)d_out, Nrows, D);
}
